// subplanar_pdd_14577119002781
// MI455X (gfx1250) — matrix operand order repaired and checked by reading, not compiled, not run on hardware
//
#include <hip/hip_runtime.h>
#include <math.h>

#define GS 29
#define NCTRL (GS * GS * GS)   // 24389
#define DW 15
#define NP (DW * DW)           // 225
#define CH3 (NP * 3)           // 675
#define NCH 8
#define V 64
#define VSTRIDE (V * V * V)    // 262144

#define KPAD 704               // 675 padded to 22 * 32
#define NTILES ((NCTRL + 15) / 16)        // 1525
#define NROWS_PAD (NTILES * 16)           // 24400

typedef __attribute__((ext_vector_type(16))) _Float16 v16h;
typedef __attribute__((ext_vector_type(8)))  _Float16 v8h;
typedef __attribute__((ext_vector_type(8)))  float    v8f;

__device__ __forceinline__ int iclamp(int v, int lo, int hi) {
    return v < lo ? lo : (v > hi ? hi : v);
}

// ---------------------------------------------------------------------------
// Stage 1: pdd[n, p, pl] = a1 + a0 * sum_c (fix[c] - mov[c])^2
// One block per control point; fix (8ch, zeros-pad trilerp) staged in LDS.
// Volumes are L2-resident (8 MB each), so the 64 gathers/sample hit L2.
// ---------------------------------------------------------------------------
__global__ __launch_bounds__(256)
void k_pdd(const float* __restrict__ f00, const float* __restrict__ f50,
           const float* __restrict__ grid_xyz, const float* __restrict__ shift,
           const float* __restrict__ alpha, float* __restrict__ pdd)
{
    __shared__ float s_fix[NCH];
    const int n = blockIdx.x;
    const float gx = grid_xyz[n * 3 + 0];
    const float gy = grid_xyz[n * 3 + 1];
    const float gz = grid_xyz[n * 3 + 2];

    // fix: trilerp of f00 at g with zeros padding; one channel per thread 0..7
    if (threadIdx.x < NCH) {
        const int c = threadIdx.x;
        float ix = 0.5f * ((gx + 1.f) * V - 1.f);
        float iy = 0.5f * ((gy + 1.f) * V - 1.f);
        float iz = 0.5f * ((gz + 1.f) * V - 1.f);
        int x0 = (int)floorf(ix), y0 = (int)floorf(iy), z0 = (int)floorf(iz);
        float fx = ix - x0, fy = iy - y0, fz = iz - z0;
        float acc = 0.f;
        #pragma unroll
        for (int dz = 0; dz < 2; ++dz) {
            float wz = dz ? fz : 1.f - fz;
            int zi = z0 + dz; int zc = iclamp(zi, 0, V - 1);
            #pragma unroll
            for (int dy = 0; dy < 2; ++dy) {
                float wy = dy ? fy : 1.f - fy;
                int yi = y0 + dy; int yc = iclamp(yi, 0, V - 1);
                #pragma unroll
                for (int dx = 0; dx < 2; ++dx) {
                    float wx = dx ? fx : 1.f - fx;
                    int xi = x0 + dx; int xc = iclamp(xi, 0, V - 1);
                    bool inb = (xi >= 0) & (xi < V) & (yi >= 0) & (yi < V) &
                               (zi >= 0) & (zi < V);
                    float w = wx * wy * wz * (inb ? 1.f : 0.f);
                    acc += w * f00[c * VSTRIDE + (zc * V + yc) * V + xc];
                }
            }
        }
        s_fix[c] = acc;
    }
    __syncthreads();

    const float a0 = alpha[0], a1 = alpha[1];
    for (int t = threadIdx.x; t < CH3; t += blockDim.x) {
        const int p = t / 3, pl = t % 3;
        const float qx = gx + shift[p * 9 + pl * 3 + 0];
        const float qy = gy + shift[p * 9 + pl * 3 + 1];
        const float qz = gz + shift[p * 9 + pl * 3 + 2];
        float ix = 0.5f * ((qx + 1.f) * V - 1.f);
        float iy = 0.5f * ((qy + 1.f) * V - 1.f);
        float iz = 0.5f * ((qz + 1.f) * V - 1.f);
        int x0 = (int)floorf(ix), y0 = (int)floorf(iy), z0 = (int)floorf(iz);
        float fx = ix - x0, fy = iy - y0, fz = iz - z0;
        float mv[NCH];
        #pragma unroll
        for (int c = 0; c < NCH; ++c) mv[c] = 0.f;
        #pragma unroll
        for (int dz = 0; dz < 2; ++dz) {
            float wz = dz ? fz : 1.f - fz;
            int zc = iclamp(z0 + dz, 0, V - 1);
            #pragma unroll
            for (int dy = 0; dy < 2; ++dy) {
                float wy = dy ? fy : 1.f - fy;
                int yc = iclamp(y0 + dy, 0, V - 1);
                #pragma unroll
                for (int dx = 0; dx < 2; ++dx) {
                    float w = wz * wy * (dx ? fx : 1.f - fx);
                    int xc = iclamp(x0 + dx, 0, V - 1);
                    const float* base = f50 + (zc * V + yc) * V + xc;
                    #pragma unroll
                    for (int c = 0; c < NCH; ++c) mv[c] += w * base[c * VSTRIDE];
                }
            }
        }
        float ssd = 0.f;
        #pragma unroll
        for (int c = 0; c < NCH; ++c) {
            float d = s_fix[c] - mv[c];
            ssd += d * d;
        }
        pdd[(size_t)n * CH3 + t] = a1 + a0 * ssd;
    }
}

// ---------------------------------------------------------------------------
// Stage 2/4: avgpool3(minpool3(replication_pad2)) over the 15x15 displacement
// plane, per (control point, plane). Entirely in LDS; one block per n.
// ---------------------------------------------------------------------------
__global__ __launch_bounds__(256)
void k_pool(const float* __restrict__ in, float* __restrict__ out)
{
    __shared__ float sX[CH3];
    __shared__ float sM[17 * 17 * 3];
    const int n = blockIdx.x;
    for (int t = threadIdx.x; t < CH3; t += blockDim.x)
        sX[t] = in[(size_t)n * CH3 + t];
    __syncthreads();
    for (int t = threadIdx.x; t < 17 * 17 * 3; t += blockDim.x) {
        const int pl = t % 3, j = (t / 3) % 17, i = t / 51;
        float m = 3.4e38f;
        #pragma unroll
        for (int u = 0; u < 3; ++u) {
            int a = iclamp(i + u - 2, 0, DW - 1);
            #pragma unroll
            for (int v = 0; v < 3; ++v) {
                int b = iclamp(j + v - 2, 0, DW - 1);
                m = fminf(m, sX[(a * DW + b) * 3 + pl]);
            }
        }
        sM[t] = m;
    }
    __syncthreads();
    for (int t = threadIdx.x; t < CH3; t += blockDim.x) {
        const int pl = t % 3, j = (t / 3) % DW, i = t / (DW * 3);
        float s = 0.f;
        #pragma unroll
        for (int u = 0; u < 3; ++u)
            #pragma unroll
            for (int v = 0; v < 3; ++v)
                s += sM[((i + u) * 17 + (j + v)) * 3 + pl];
        out[(size_t)n * CH3 + t] = s * (1.f / 9.f);
    }
}

// ---------------------------------------------------------------------------
// grid_smooth = two 3x3 avg pools with replication pad over control-grid
// (d, h)  ==  separable 5-tap [1,2,3,2,1]/9 filter with clamped indexing.
// Split into a d-pass and an h-pass; h-pass optionally fuses
// cost = a4 + a2*pdd + a3*smooth.
// ---------------------------------------------------------------------------
__global__ __launch_bounds__(256)
void k_smooth_d(const float* __restrict__ in, float* __restrict__ out)
{
    const int total = NCTRL * CH3;
    const float K[5] = {1.f, 2.f, 3.f, 2.f, 1.f};
    for (int idx = blockIdx.x * blockDim.x + threadIdx.x; idx < total;
         idx += gridDim.x * blockDim.x) {
        const int c = idx % CH3;
        const int n = idx / CH3;
        const int w = n % GS, h = (n / GS) % GS, d = n / (GS * GS);
        float s = 0.f;
        #pragma unroll
        for (int t = 0; t < 5; ++t) {
            int dd = iclamp(d + t - 2, 0, GS - 1);
            s += K[t] * in[((size_t)(dd * GS * GS + h * GS + w)) * CH3 + c];
        }
        out[idx] = s * (1.f / 9.f);
    }
}

__global__ __launch_bounds__(256)
void k_smooth_h(const float* __restrict__ in, float* __restrict__ out,
                const float* __restrict__ pdd, const float* __restrict__ alpha,
                int combine)
{
    const int total = NCTRL * CH3;
    const float K[5] = {1.f, 2.f, 3.f, 2.f, 1.f};
    for (int idx = blockIdx.x * blockDim.x + threadIdx.x; idx < total;
         idx += gridDim.x * blockDim.x) {
        const int c = idx % CH3;
        const int n = idx / CH3;
        const int w = n % GS, h = (n / GS) % GS, d = n / (GS * GS);
        float s = 0.f;
        #pragma unroll
        for (int t = 0; t < 5; ++t) {
            int hh = iclamp(h + t - 2, 0, GS - 1);
            s += K[t] * in[((size_t)(d * GS * GS + hh * GS + w)) * CH3 + c];
        }
        s *= (1.f / 9.f);
        if (combine)
            out[idx] = alpha[4] + alpha[2] * pdd[idx] + alpha[3] * s;
        else
            out[idx] = s;
    }
}

// ---------------------------------------------------------------------------
// Softmax over the 225 displacements per (n, plane). One wave32 per (n, pl).
// Also emits the f16 A-panel (row pitch KPAD, zero k-padding) for the
// WMMA prediction GEMM, so the GEMM loop needs no bounds checks or converts.
// ---------------------------------------------------------------------------
__global__ __launch_bounds__(32)
void k_softmax(const float* __restrict__ cost_avg,
               const float* __restrict__ alpha, float* __restrict__ cost_soft,
               _Float16* __restrict__ Ah)
{
    const int b = blockIdx.x;
    const int n = b / 3, pl = b % 3, lane = threadIdx.x;
    const float a5 = alpha[5];
    float vals[8];
    float m = -3.4e38f;
    #pragma unroll
    for (int i = 0; i < 8; ++i) {
        int p = lane + 32 * i;
        float t = (p < NP) ? -a5 * cost_avg[(size_t)n * CH3 + p * 3 + pl]
                           : -3.4e38f;
        vals[i] = t;
        m = fmaxf(m, t);
    }
    for (int off = 16; off >= 1; off >>= 1)
        m = fmaxf(m, __shfl_xor(m, off, 32));
    float s = 0.f;
    #pragma unroll
    for (int i = 0; i < 8; ++i) {
        int p = lane + 32 * i;
        float e = (p < NP) ? __expf(vals[i] - m) : 0.f;
        vals[i] = e;
        s += e;
    }
    for (int off = 16; off >= 1; off >>= 1)
        s += __shfl_xor(s, off, 32);
    const float inv = 1.f / s;
    #pragma unroll
    for (int i = 0; i < 8; ++i) {
        int p = lane + 32 * i;
        if (p < NP) {
            float v = vals[i] * inv;
            cost_soft[(size_t)n * CH3 + p * 3 + pl] = v;
            Ah[(size_t)n * KPAD + p * 3 + pl] = (_Float16)v;
        }
    }
    // zero the K-padding of this row once (by the pl==0 block)
    if (pl == 0 && lane < (KPAD - CH3))
        Ah[(size_t)n * KPAD + CH3 + lane] = (_Float16)0.f;
}

// Zero the padded tail rows of the A-panel (rows NCTRL .. NROWS_PAD-1).
__global__ __launch_bounds__(256)
void k_zero_tail(_Float16* __restrict__ Ah)
{
    const int total = (NROWS_PAD - NCTRL) * KPAD;
    for (int i = blockIdx.x * blockDim.x + threadIdx.x; i < total;
         i += gridDim.x * blockDim.x)
        Ah[(size_t)NCTRL * KPAD + i] = (_Float16)0.f;
}

// ---------------------------------------------------------------------------
// pred_xyz = 0.5 * (cost_soft (N x 675)) @ (shift (675 x 3)) as a real GEMM
// via v_wmma_f32_16x16x32_f16: one wave per 16 control points, K padded to 704.
// A layout (16-bit A 16x32): M = lane%16; e -> K = e + 8*(e>=8) + 8*(lane>=16)
//   => two aligned 16B loads from the padded f16 panel per K-step.
// B layout (16-bit B 32x16): N = lane%16; e -> K = e + 8*(e>=8) + 8*(lane>=16)
//   => two aligned 16B LDS loads from a per-block staged f16 shift panel.
// D layout (f32 C/D 16x16):  col = lane%16; vgpr r -> row = r + 8*(lane>=16)
// ---------------------------------------------------------------------------
__global__ __launch_bounds__(256)
void k_pred_wmma(const _Float16* __restrict__ Ah,
                 const float* __restrict__ shift, float* __restrict__ pred)
{
    __shared__ _Float16 sB[16 * KPAD];    // sB[n][k], zero outside n<3 / k<675
    for (int t = threadIdx.x; t < 16 * KPAD; t += blockDim.x) {
        const int nn = t / KPAD, kk = t % KPAD;
        float bv = 0.f;
        if (nn < 3 && kk < CH3) {         // k = p*3 + pl ; shift[p, pl, d]
            const int p = kk / 3, pl = kk % 3;
            bv = shift[p * 9 + pl * 3 + nn];
        }
        sB[t] = (_Float16)bv;
    }
    __syncthreads();

    const int wave = (blockIdx.x * blockDim.x + threadIdx.x) >> 5;
    const int lane = threadIdx.x & 31;
    if (wave >= NTILES) return;           // wave-uniform: EXEC stays all-ones

    const int n0 = wave * 16;
    const int m = lane & 15;
    const int grp = lane >> 4;            // 0 or 1

    const _Float16* rowA = Ah + (size_t)(n0 + m) * KPAD + grp * 8;
    const _Float16* rowB = sB + m * KPAD + grp * 8;
    v8f acc = {};

    for (int k0 = 0; k0 < KPAD; k0 += 32) {
        v8h a0 = *(const v8h*)(rowA + k0);        // e = 0..7
        v8h a1 = *(const v8h*)(rowA + k0 + 16);   // e = 8..15
        v8h b0 = *(const v8h*)(rowB + k0);        // e = 0..7
        v8h b1 = *(const v8h*)(rowB + k0 + 16);   // e = 8..15
        v16h a, b;
        #pragma unroll
        for (int e = 0; e < 8; ++e) {
            a[e] = a0[e]; a[e + 8] = a1[e];
            b[e] = b0[e]; b[e + 8] = b1[e];
        }
        acc = __builtin_amdgcn_wmma_f32_16x16x32_f16(
            false, a, false, b, (short)0, acc, false, false);
    }

    if (m < 3) {
        #pragma unroll
        for (int r = 0; r < 8; ++r) {
            const int row = n0 + r + grp * 8;
            if (row < NCTRL) pred[(size_t)row * 3 + m] = 0.5f * acc[r];
        }
    }
}

// ---------------------------------------------------------------------------
extern "C" void kernel_launch(void* const* d_in, const int* in_sizes, int n_in,
                              void* d_out, int out_size, void* d_ws,
                              size_t ws_size, hipStream_t stream)
{
    const float* f00      = (const float*)d_in[0];   // (1,8,64,64,64)
    const float* f50      = (const float*)d_in[1];   // (1,8,64,64,64)
    /* d_in[2] = shift_2d_min (broadcast copy, unused) */
    const float* grid_xyz = (const float*)d_in[3];   // (N,3)
    const float* shift    = (const float*)d_in[4];   // (P,3,3)
    const float* alpha    = (const float*)d_in[5];   // (6,)

    float* out_soft = (float*)d_out;                       // N*675
    float* out_pred = out_soft + (size_t)NCTRL * CH3;      // N*3
    float* out_avg  = out_pred + (size_t)NCTRL * 3;        // N*675

    float* B0 = (float*)d_ws;                              // N*675 floats
    float* B1 = B0 + (size_t)NCTRL * CH3;                  // N*675 floats
    _Float16* Ah = (_Float16*)B0;   // f16 A-panel (24400 x 704) reuses B0
    (void)ws_size; (void)in_sizes; (void)n_in; (void)out_size;

    const int EB = 2048;   // elementwise grid (grid-stride)

    // 1) pdd -> B0
    k_pdd<<<NCTRL, 256, 0, stream>>>(f00, f50, grid_xyz, shift, alpha, B0);
    // 2) cost = pool(pdd) -> B1
    k_pool<<<NCTRL, 256, 0, stream>>>(B0, B1);
    // 3) cost = a4 + a2*pdd + a3*grid_smooth(cost) -> B1
    //    (out_avg region doubles as scratch for the d-pass; rewritten later)
    k_smooth_d<<<EB, 256, 0, stream>>>(B1, out_avg);
    k_smooth_h<<<EB, 256, 0, stream>>>(out_avg, B1, B0, alpha, 1);
    // 4) cost = pool(cost) -> B0  (pdd dead now)
    k_pool<<<NCTRL, 256, 0, stream>>>(B1, B0);
    // 5) cost_avg = grid_smooth(cost) -> out_avg
    k_smooth_d<<<EB, 256, 0, stream>>>(B0, B1);
    k_smooth_h<<<EB, 256, 0, stream>>>(B1, out_avg, B1, alpha, 0);
    // 6) cost_soft = softmax(-a5 * cost_avg); also emits f16 A-panel into B0
    //    (B0 is dead as f32 scratch after stage 5's d-pass)
    k_softmax<<<NCTRL * 3, 32, 0, stream>>>(out_avg, alpha, out_soft, Ah);
    k_zero_tail<<<31, 256, 0, stream>>>(Ah);
    // 7) pred_xyz = 0.5 * cost_soft @ shift  (WMMA GEMM)
    const int blocks = (NTILES * 32 + 255) / 256;
    k_pred_wmma<<<blocks, 256, 0, stream>>>(Ah, shift, out_pred);
}
